// GraphEmbeddings_60971355734503
// MI455X (gfx1250) — matrix operand order repaired and checked by reading, not compiled, not run on hardware
//
#include <hip/hip_runtime.h>
#include <hip/hip_fp16.h>

// ---------------------------------------------------------------------------
// GATv2 x2 + BN + attentional pooling for gfx1250 (MI455X).
// Dense node transforms run on v_wmma_f32_16x16x32_f16 with pre-packed
// fragment-order weights (contiguous 32B B-fragment loads); edge stages are
// wave32-per-edge streaming kernels with L2-resident node features and
// LDS-staged edge-weight matrices.
// ---------------------------------------------------------------------------

typedef __attribute__((ext_vector_type(16))) _Float16 v16h;
typedef __attribute__((ext_vector_type(8)))  _Float16 v8h;
typedef __attribute__((ext_vector_type(8)))  float    v8f;

#define NEG_ATT 0.2f
#define NEG_ACT 0.01f
#define BN_EPS  1e-5f
#define ENC_NEG_INF 0x007FFFFFu   // monotonic encoding of -inf

// monotonic float <-> uint mapping (order preserving, incl. negatives)
__device__ __forceinline__ unsigned fenc(float f) {
  unsigned u = __float_as_uint(f);
  return (u & 0x80000000u) ? ~u : (u | 0x80000000u);
}
__device__ __forceinline__ float fdec(unsigned u) {
  return __uint_as_float((u & 0x80000000u) ? (u ^ 0x80000000u) : ~u);
}

// ---------------------------------------------------------------- fills
__global__ void fill_f32(float* p, float v, int n) {
  int i = blockIdx.x * blockDim.x + threadIdx.x;
  if (i < n) p[i] = v;
}
__global__ void fill_u32(unsigned* p, unsigned v, int n) {
  int i = blockIdx.x * blockDim.x + threadIdx.x;
  if (i < n) p[i] = v;
}
__global__ void rows_bias(float* out, const float* __restrict__ bias, int total, int C) {
  int i = blockIdx.x * blockDim.x + threadIdx.x;
  if (i < total) out[i] = bias[i % C];
}

// --------------------------------------------------- weight pack (B fragment)
// Pack row-major f32 B[K,N] into WMMA B-fragment order, f16:
//   P[((k0/32)*(N/16) + nTile)*512 + lane*16 + h] = B[k, n]
//   with k = k0 + (lane>>4)*16 + h, n = nTile*16 + (lane&15)
// so each lane's 16 halves for one fragment are 32 contiguous bytes.
__global__ void pack_b(const float* __restrict__ B, _Float16* __restrict__ P,
                       int K, int N) {
  int i = blockIdx.x * blockDim.x + threadIdx.x;
  if (i >= K * N) return;
  int h    = i & 15;
  int lane = (i >> 4) & 31;
  int blkf = i >> 9;              // fragment index = (k0/32)*(N/16) + nTile
  int nT   = blkf % (N >> 4);
  int k0   = (blkf / (N >> 4)) << 5;
  int k = k0 + ((lane >> 4) << 4) + h;
  int n = (nT << 4) + (lane & 15);
  P[i] = (_Float16)B[(size_t)k * N + n];
}

// ------------------------------------------------- embedding (argmax gather)
__global__ void embed_kernel(const float* __restrict__ x, const float* __restrict__ tab,
                             _Float16* __restrict__ h16, int NN) {
  int n = blockIdx.x * blockDim.x + threadIdx.x;
  if (n >= NN) return;
  const float* xr = x + (size_t)n * 64;
  float best = xr[0]; int bi = 0;
  #pragma unroll 4
  for (int i = 1; i < 64; i++) { float v = xr[i]; if (v > best) { best = v; bi = i; } }
  const float* er = tab + (size_t)bi * 128;
  _Float16* hr = h16 + (size_t)n * 128;
  #pragma unroll 4
  for (int i = 0; i < 128; i++) hr[i] = (_Float16)er[i];
}

// ---------------------------------------- self-loop edge_attr mean (PyG 'mean')
__global__ void loop_accum(const int* __restrict__ ei, const float* __restrict__ ea,
                           float* deg, float* easum, int NE) {
  int e = blockIdx.x * blockDim.x + threadIdx.x;
  if (e >= NE) return;
  int dst = ei[NE + e];
  atomicAdd(deg + dst, 1.0f);
  const float* row = ea + (size_t)e * 16;
  float* out = easum + (size_t)dst * 16;
  #pragma unroll
  for (int k = 0; k < 16; k++) atomicAdd(out + k, row[k]);
}
__global__ void loop_div(float* easum, const float* __restrict__ deg, int NN) {
  int i = blockIdx.x * blockDim.x + threadIdx.x;
  if (i >= NN * 16) return;
  float d = fmaxf(deg[i >> 4], 1.0f);
  easum[i] /= d;
}

// -------------------------------------------------------- WMMA f16 GEMM
// C[M,N] = A[M,K]@Bpacked + bias. Wave computes two adjacent 16x16 tiles
// (reusing the A fragment), K-step 32. 128-thread block covers 128 columns.
// A 16x32 f16 fragment (matches packed B): lane<16 -> M=lane,
//   halves[0..15]=K(k0..k0+15); lane>=16: K +16.
__global__ void gemm_wmma(const _Float16* __restrict__ A, const _Float16* __restrict__ Bp,
                          const float* __restrict__ bias, float* __restrict__ C,
                          int M, int K, int N) {
  int lane  = threadIdx.x & 31;
  int wave  = threadIdx.x >> 5;
  int mTile = blockIdx.x;
  int nTile0 = blockIdx.y * 8 + wave * 2;   // two n-tiles per wave
  if (mTile * 16 >= M) return;
  int lo = lane & 15, hi = lane >> 4;
  int m = mTile * 16 + lo;
  const _Float16* arow = A + (size_t)m * K;
  int nFrag = N >> 4;
  v8f acc0 = {}, acc1 = {};
  for (int k0 = 0; k0 < K; k0 += 32) {
    v8h a0 = *(const v8h*)(arow + k0 + hi * 16);
    v8h a1 = *(const v8h*)(arow + k0 + hi * 16 + 8);
    v16h a;
    #pragma unroll
    for (int i = 0; i < 8; i++) { a[i] = a0[i]; a[i + 8] = a1[i]; }
    const _Float16* bbase = Bp + ((size_t)(k0 >> 5) * nFrag + nTile0) * 512 + lane * 16;
    v16h b0 = *(const v16h*)(bbase);
    v16h b1 = *(const v16h*)(bbase + 512);
    acc0 = __builtin_amdgcn_wmma_f32_16x16x32_f16(false, a, false, b0,
                                                  (short)0, acc0, false, false);
    acc1 = __builtin_amdgcn_wmma_f32_16x16x32_f16(false, a, false, b1,
                                                  (short)0, acc1, false, false);
  }
  // C/D layout: lane gives column; vgpr r -> row = 16*mTile + hi*8 + r.
  int rbase = mTile * 16 + hi * 8;
  int n0 = nTile0 * 16 + lo;
  float bs0 = bias[n0], bs1 = bias[n0 + 16];
  #pragma unroll
  for (int r = 0; r < 8; r++) {
    float* crow = C + (size_t)(rbase + r) * N + n0;
    crow[0]  = acc0[r] + bs0;
    crow[16] = acc1[r] + bs1;
  }
}

// ---------------------------------------------- GATv2 edge passes (wave/edge)
template <int C>
__global__ void edge_logits(const int* __restrict__ ei, const float* __restrict__ ea_in,
                            const float* __restrict__ ea_mean,
                            const float* __restrict__ xl, const float* __restrict__ xr,
                            const float* __restrict__ We, const float* __restrict__ att,
                            float* __restrict__ logit, unsigned* mmax, int NE, int NN) {
  __shared__ float sWe[16 * C];
  __shared__ float sAtt[C];
  // cooperative LDS staging (before any early exit: uniform __syncthreads)
  for (int i = threadIdx.x; i < 16 * C; i += blockDim.x) sWe[i] = We[i];
  for (int i = threadIdx.x; i < C; i += blockDim.x) sAtt[i] = att[i];
  __syncthreads();

  int gw = blockIdx.x * (blockDim.x >> 5) + (threadIdx.x >> 5);
  int E2 = NE + NN;
  if (gw >= E2) return;
  int lane = threadIdx.x & 31;
  int src, dst; const float* ea;
  if (gw < NE) { src = ei[gw]; dst = ei[NE + gw]; ea = ea_in + (size_t)gw * 16; }
  else         { src = dst = gw - NE;             ea = ea_mean + (size_t)(gw - NE) * 16; }
  float eav = (lane < 16) ? ea[lane] : 0.0f;       // 16 edge feats, shfl-broadcast
  const float* xls = xl + (size_t)src * C;
  const float* xrd = xr + (size_t)dst * C;
  float acc = 0.0f;
  #pragma unroll
  for (int j = 0; j < C / 32; j++) {
    int c = j * 32 + lane;
    float w = 0.0f;
    #pragma unroll
    for (int k = 0; k < 16; k++) w += __shfl(eav, k) * sWe[k * C + c];
    float v = xls[c] + xrd[c] + w;
    v = v > 0.0f ? v : NEG_ATT * v;                 // attention LeakyReLU(0.2)
    acc += v * sAtt[c];
  }
  #pragma unroll
  for (int off = 16; off > 0; off >>= 1) acc += __shfl_xor(acc, off);
  if (lane == 0) {
    logit[gw] = acc;
    atomicMax(mmax + dst, fenc(acc));
  }
}

__global__ void edge_softmax_den(const int* __restrict__ ei, const float* __restrict__ logit,
                                 float* __restrict__ exv, const unsigned* __restrict__ mmax,
                                 float* denom, int NE, int NN) {
  int e = blockIdx.x * blockDim.x + threadIdx.x;
  int E2 = NE + NN;
  if (e >= E2) return;
  int dst = (e < NE) ? ei[NE + e] : (e - NE);
  float ex = __expf(logit[e] - fdec(mmax[dst]));
  exv[e] = ex;
  atomicAdd(denom + dst, ex);
}

template <int C>
__global__ void edge_aggregate(const int* __restrict__ ei, const float* __restrict__ exv,
                               const float* __restrict__ denom, const float* __restrict__ xl,
                               float* out, int NE, int NN) {
  int gw = blockIdx.x * (blockDim.x >> 5) + (threadIdx.x >> 5);
  int E2 = NE + NN;
  if (gw >= E2) return;
  int lane = threadIdx.x & 31;
  int src, dst;
  if (gw < NE) { src = ei[gw]; dst = ei[NE + gw]; }
  else         { src = dst = gw - NE; }
  float alpha = exv[gw] / denom[dst];
  const float* xls = xl + (size_t)src * C;
  float* od = out + (size_t)dst * C;
  #pragma unroll
  for (int j = 0; j < C / 32; j++) {
    int c = j * 32 + lane;
    atomicAdd(od + c, xls[c] * alpha);
  }
}

// ----------------------------------------------------------- BatchNorm
__global__ void bn_stats(const float* __restrict__ x, float* mu, float* rs, int NN, int C) {
  int c = blockIdx.x, t = threadIdx.x;
  float s = 0.0f, s2 = 0.0f;
  for (int n = t; n < NN; n += blockDim.x) {
    float v = x[(size_t)n * C + c];
    s += v; s2 += v * v;
  }
  __shared__ float sh1[256], sh2[256];
  sh1[t] = s; sh2[t] = s2; __syncthreads();
  for (int o = 128; o > 0; o >>= 1) {
    if (t < o) { sh1[t] += sh1[t + o]; sh2[t] += sh2[t + o]; }
    __syncthreads();
  }
  if (t == 0) {
    float m = sh1[0] / (float)NN;
    float var = sh2[0] / (float)NN - m * m;   // biased, as torch normalizes
    mu[c] = m;
    rs[c] = rsqrtf(var + BN_EPS);
  }
}
__global__ void bn_apply(float* x, const float* __restrict__ mu, const float* __restrict__ rs,
                         const float* __restrict__ g, const float* __restrict__ b,
                         _Float16* xh, int total, int C) {
  int i = blockIdx.x * blockDim.x + threadIdx.x;
  if (i >= total) return;
  int c = i % C;
  float v = (x[i] - mu[c]) * rs[c] * g[c] + b[c];
  v = v > 0.0f ? v : NEG_ACT * v;             // activation LeakyReLU(0.01)
  x[i] = v;
  if (xh) xh[i] = (_Float16)v;
}

// --------------------------------------------- gate + attentional pooling
__global__ void gate_kernel(const float* __restrict__ d2, const float* __restrict__ Wg,
                            const float* __restrict__ bg, const int* __restrict__ batch,
                            float* gate, unsigned* gmax, int NN) {
  int n = blockIdx.x * blockDim.x + threadIdx.x;
  if (n >= NN) return;
  const float* r = d2 + (size_t)n * 256;
  float s = bg[0];
  #pragma unroll 4
  for (int c = 0; c < 256; c++) s += r[c] * Wg[c];
  gate[n] = s;
  atomicMax(gmax + batch[n], fenc(s));
}
__global__ void gate_exp(const float* __restrict__ gate, const unsigned* __restrict__ gmax,
                         const int* __restrict__ batch, float* gev, float* gden, int NN) {
  int n = blockIdx.x * blockDim.x + threadIdx.x;
  if (n >= NN) return;
  float e = __expf(gate[n] - fdec(gmax[batch[n]]));
  gev[n] = e;
  atomicAdd(gden + batch[n], e);
}
__global__ void agg_kernel(const float* __restrict__ d2, const float* __restrict__ gev,
                           const float* __restrict__ gden, const int* __restrict__ batch,
                           float* out, int NN) {
  int gw = blockIdx.x * (blockDim.x >> 5) + (threadIdx.x >> 5);
  if (gw >= NN) return;
  int lane = threadIdx.x & 31;
  int b = batch[gw];
  float d = gden[b];
  if (d == 0.0f) d = 1.0f;
  float w = gev[gw] / d;
  const float* r = d2 + (size_t)gw * 256;
  float* o = out + (size_t)b * 256;
  #pragma unroll
  for (int j = 0; j < 8; j++) {
    int c = j * 32 + lane;
    atomicAdd(o + c, r[c] * w);
  }
}

// ---------------------------------------------------------------------------
extern "C" void kernel_launch(void* const* d_in, const int* in_sizes, int n_in,
                              void* d_out, int out_size, void* d_ws, size_t ws_size,
                              hipStream_t stream) {
  (void)n_in; (void)out_size; (void)ws_size;
  const float* x     = (const float*)d_in[0];
  const int*   ei    = (const int*)  d_in[1];
  const float* ea    = (const float*)d_in[2];
  const int*   batch = (const int*)  d_in[3];
  const float* emb   = (const float*)d_in[4];
  const float* Wl1   = (const float*)d_in[5];
  const float* bl1   = (const float*)d_in[6];
  const float* Wr1   = (const float*)d_in[7];
  const float* br1   = (const float*)d_in[8];
  const float* We1   = (const float*)d_in[9];
  const float* att1  = (const float*)d_in[10];
  const float* bias1 = (const float*)d_in[11];
  const float* g1    = (const float*)d_in[12];
  const float* be1   = (const float*)d_in[13];
  const float* Wl2   = (const float*)d_in[14];
  const float* bl2   = (const float*)d_in[15];
  const float* Wr2   = (const float*)d_in[16];
  const float* br2   = (const float*)d_in[17];
  const float* We2   = (const float*)d_in[18];
  const float* att2  = (const float*)d_in[19];
  const float* bias2 = (const float*)d_in[20];
  const float* g2    = (const float*)d_in[21];
  const float* be2   = (const float*)d_in[22];
  const float* Wg    = (const float*)d_in[23];
  const float* bg    = (const float*)d_in[24];

  const int NN = in_sizes[0] / 64;   // 50000
  const int NE = in_sizes[1] / 2;    // 800000
  const int E2 = NE + NN;            // with self-loops

  // ---- workspace carve-out
  char* base = (char*)d_ws;
  size_t off = 0;
  auto take = [&](size_t bytes) -> void* {
    void* p = base + off;
    off = (off + bytes + 255) & ~(size_t)255;
    return p;
  };
  _Float16* h16   = (_Float16*)take((size_t)NN * 128 * 2);
  _Float16* wl1p  = (_Float16*)take(128 * 128 * 2);
  _Float16* wr1p  = (_Float16*)take(128 * 128 * 2);
  _Float16* wl2p  = (_Float16*)take(128 * 256 * 2);
  _Float16* wr2p  = (_Float16*)take(128 * 256 * 2);
  float* xl    = (float*)take((size_t)NN * 256 * 4);
  float* xr    = (float*)take((size_t)NN * 256 * 4);
  float* d1    = (float*)take((size_t)NN * 128 * 4);
  _Float16* d1h = (_Float16*)take((size_t)NN * 128 * 2);
  float* d2    = (float*)take((size_t)NN * 256 * 4);
  float* deg   = (float*)take((size_t)NN * 4);
  float* eam   = (float*)take((size_t)NN * 16 * 4);
  float* logit = (float*)take((size_t)E2 * 4);
  float* exv   = (float*)take((size_t)E2 * 4);
  unsigned* mmax = (unsigned*)take((size_t)NN * 4);
  float* denom = (float*)take((size_t)NN * 4);
  float* mu    = (float*)take(256 * 4);
  float* rs    = (float*)take(256 * 4);
  float* gate  = (float*)take((size_t)NN * 4);
  float* gev   = (float*)take((size_t)NN * 4);
  unsigned* gmax = (unsigned*)take(256 * 4);
  float* gden  = (float*)take(256 * 4);

  auto blk = [](int n, int t) { return (n + t - 1) / t; };

  // ---- pack weights into WMMA B-fragment order (f16)
  pack_b<<<blk(128 * 128, 256), 256, 0, stream>>>(Wl1, wl1p, 128, 128);
  pack_b<<<blk(128 * 128, 256), 256, 0, stream>>>(Wr1, wr1p, 128, 128);
  pack_b<<<blk(128 * 256, 256), 256, 0, stream>>>(Wl2, wl2p, 128, 256);
  pack_b<<<blk(128 * 256, 256), 256, 0, stream>>>(Wr2, wr2p, 128, 256);

  // ---- embedding lookup
  embed_kernel<<<blk(NN, 256), 256, 0, stream>>>(x, emb, h16, NN);

  // ---- self-loop edge_attr mean
  fill_f32<<<blk(NN, 256), 256, 0, stream>>>(deg, 0.0f, NN);
  fill_f32<<<blk(NN * 16, 256), 256, 0, stream>>>(eam, 0.0f, NN * 16);
  loop_accum<<<blk(NE, 256), 256, 0, stream>>>(ei, ea, deg, eam, NE);
  loop_div<<<blk(NN * 16, 256), 256, 0, stream>>>(eam, deg, NN);

  // ============================ layer 1 (C=128) ============================
  {
    dim3 g((NN + 15) / 16, 128 / 128);   // block covers 128 columns
    gemm_wmma<<<g, 128, 0, stream>>>(h16, wl1p, bl1, xl, NN, 128, 128);
    gemm_wmma<<<g, 128, 0, stream>>>(h16, wr1p, br1, xr, NN, 128, 128);
  }
  fill_u32<<<blk(NN, 256), 256, 0, stream>>>(mmax, ENC_NEG_INF, NN);
  fill_f32<<<blk(NN, 256), 256, 0, stream>>>(denom, 0.0f, NN);
  rows_bias<<<blk(NN * 128, 256), 256, 0, stream>>>(d1, bias1, NN * 128, 128);

  edge_logits<128><<<blk(E2, 8), 256, 0, stream>>>(ei, ea, eam, xl, xr, We1, att1,
                                                   logit, mmax, NE, NN);
  edge_softmax_den<<<blk(E2, 256), 256, 0, stream>>>(ei, logit, exv, mmax, denom, NE, NN);
  edge_aggregate<128><<<blk(E2, 8), 256, 0, stream>>>(ei, exv, denom, xl, d1, NE, NN);

  bn_stats<<<128, 256, 0, stream>>>(d1, mu, rs, NN, 128);
  bn_apply<<<blk(NN * 128, 256), 256, 0, stream>>>(d1, mu, rs, g1, be1, d1h, NN * 128, 128);

  // ============================ layer 2 (C=256) ============================
  {
    dim3 g((NN + 15) / 16, 256 / 128);
    gemm_wmma<<<g, 128, 0, stream>>>(d1h, wl2p, bl2, xl, NN, 128, 256);
    gemm_wmma<<<g, 128, 0, stream>>>(d1h, wr2p, br2, xr, NN, 128, 256);
  }
  fill_u32<<<blk(NN, 256), 256, 0, stream>>>(mmax, ENC_NEG_INF, NN);
  fill_f32<<<blk(NN, 256), 256, 0, stream>>>(denom, 0.0f, NN);
  rows_bias<<<blk(NN * 256, 256), 256, 0, stream>>>(d2, bias2, NN * 256, 256);

  edge_logits<256><<<blk(E2, 8), 256, 0, stream>>>(ei, ea, eam, xl, xr, We2, att2,
                                                   logit, mmax, NE, NN);
  edge_softmax_den<<<blk(E2, 256), 256, 0, stream>>>(ei, logit, exv, mmax, denom, NE, NN);
  edge_aggregate<256><<<blk(E2, 8), 256, 0, stream>>>(ei, exv, denom, xl, d2, NE, NN);

  bn_stats<<<256, 256, 0, stream>>>(d2, mu, rs, NN, 256);
  bn_apply<<<blk(NN * 256, 256), 256, 0, stream>>>(d2, mu, rs, g2, be2, nullptr, NN * 256, 256);

  // ===================== attentional aggregation to d_out ==================
  fill_u32<<<1, 256, 0, stream>>>(gmax, ENC_NEG_INF, 256);
  fill_f32<<<1, 256, 0, stream>>>(gden, 0.0f, 256);
  fill_f32<<<blk(256 * 256, 256), 256, 0, stream>>>((float*)d_out, 0.0f, 256 * 256);

  gate_kernel<<<blk(NN, 256), 256, 0, stream>>>(d2, Wg, bg, batch, gate, gmax, NN);
  gate_exp<<<blk(NN, 256), 256, 0, stream>>>(gate, gmax, batch, gev, gden, NN);
  agg_kernel<<<blk(NN, 8), 256, 0, stream>>>(d2, gev, gden, batch, (float*)d_out, NN);
}
